// Griffin_24180665877251
// MI455X (gfx1250) — matrix operand order repaired and checked by reading, not compiled, not run on hardware
//
#include <hip/hip_runtime.h>
#include <hip/hip_bf16.h>
#include <math.h>
#include <stdint.h>

typedef __bf16 bf16_t;
typedef __attribute__((ext_vector_type(16))) __bf16 v16bf;
typedef __attribute__((ext_vector_type(8)))  __bf16 v8bf;
typedef __attribute__((ext_vector_type(8)))  float  v8f;

constexpr int kDim = 1024, kHid = 1536, kH2 = 3072, kGhid = 2048, kG2 = 4096;
constexpr int kBatch = 4, kT = 2048, kNT = 8192;
constexpr int kNCh = 16, kCL = kT / kNCh;             // chunked scan: 16 chunks of 128
constexpr int BM = 128, BN = 128, BK = 32, LDSS = 40; // LDS row stride (bf16), 80B, 16B aligned
constexpr unsigned kBufBytes = BM * LDSS * 2;         // 10240 B per LDS tile buffer

// CDNA5 async global->LDS copy (ASYNCcnt-tracked). VDST = per-lane LDS byte
// offset, VADDR = per-lane 64-bit global address. 16 bytes per lane.
__device__ __forceinline__ void async_copy_b128(unsigned lds_byte, const void* gaddr) {
  asm volatile("global_load_async_to_lds_b128 %0, %1, off"
               :: "v"(lds_byte), "v"((unsigned long long)(uintptr_t)gaddr)
               : "memory");
}
__device__ __forceinline__ void wait_async0() {
  asm volatile("s_wait_asynccnt 0x0" ::: "memory");
}

// ---------------------------------------------------------------------------
// Tiled bf16 WMMA GEMM: C[M,N] (f32) = A[M,K] (bf16, row-major) * W[N,K]^T (bf16)
// Double-buffered LDS with async global->LDS copies; one barrier per K-step.
// Block = 256 threads = 8 waves; wave tile 32x64; block tile 128x128x(K/32).
// Optional fused bias[n] and residual[m,n] adds in the epilogue.
// ---------------------------------------------------------------------------
__global__ __launch_bounds__(256) void gemm_bf16(
    const bf16_t* __restrict__ A, const bf16_t* __restrict__ W,
    float* __restrict__ C, int M, int N, int K,
    const float* __restrict__ bias, const float* __restrict__ residual)
{
  __shared__ __align__(16) bf16_t As[2][BM * LDSS];
  __shared__ __align__(16) bf16_t Bs[2][BN * LDSS];
  const int tid   = threadIdx.x;
  const int lane  = tid & 31;
  const int wave  = tid >> 5;
  const int waveM = wave & 3;    // 4 waves along M
  const int waveN = wave >> 2;   // 2 waves along N
  const int laneM = lane & 15;
  const int laneK = lane >> 4;   // 0 or 1
  const size_t blockM = (size_t)blockIdx.y * BM;
  const size_t blockN = (size_t)blockIdx.x * BN;

  // Per-thread copy slots: two 16B chunks each for A and B (8 KB tile / 256 thr)
  const int e0 = tid * 8;
  const int r0 = e0 >> 5, c0 = e0 & 31;        // rows 0..63
  const int e1 = (256 + tid) * 8;
  const int r1 = e1 >> 5, c1 = e1 & 31;        // rows 64..127
  const bf16_t* gA0 = A + (blockM + r0) * (size_t)K + c0;
  const bf16_t* gA1 = A + (blockM + r1) * (size_t)K + c1;
  const bf16_t* gB0 = W + (blockN + r0) * (size_t)K + c0;
  const bf16_t* gB1 = W + (blockN + r1) * (size_t)K + c1;
  const unsigned lA0 = (unsigned)(uintptr_t)&As[0][r0 * LDSS + c0];
  const unsigned lA1 = (unsigned)(uintptr_t)&As[0][r1 * LDSS + c1];
  const unsigned lB0 = (unsigned)(uintptr_t)&Bs[0][r0 * LDSS + c0];
  const unsigned lB1 = (unsigned)(uintptr_t)&Bs[0][r1 * LDSS + c1];

  v8f acc[2][4];
#pragma unroll
  for (int i = 0; i < 2; ++i)
#pragma unroll
    for (int j = 0; j < 4; ++j)
#pragma unroll
      for (int v = 0; v < 8; ++v) acc[i][j][v] = 0.0f;

  // Prologue: tile 0 -> buffer 0
  async_copy_b128(lA0, gA0);
  async_copy_b128(lA1, gA1);
  async_copy_b128(lB0, gB0);
  async_copy_b128(lB1, gB1);

  int cur = 0;
  for (int k0 = 0; k0 < K; k0 += BK) {
    wait_async0();        // this wave's slice of tile k0 is in LDS
    __syncthreads();      // every wave's slice is in LDS; prev reads of buf^1 done

    if (k0 + BK < K) {    // issue tile k0+BK into the other buffer
      const unsigned d = (unsigned)(cur ^ 1) * kBufBytes;
      async_copy_b128(lA0 + d, gA0 + k0 + BK);
      async_copy_b128(lA1 + d, gA1 + k0 + BK);
      async_copy_b128(lB0 + d, gB0 + k0 + BK);
      async_copy_b128(lB1 + d, gB1 + k0 + BK);
    }

    const bf16_t* as = As[cur];
    const bf16_t* bs = Bs[cur];
    // A fragment 16x32: lane<16 holds K {0..7,16..23}, lane>=16 holds {8..15,24..31}
    v16bf aF[2], bF[4];
#pragma unroll
    for (int tm = 0; tm < 2; ++tm) {
      const bf16_t* p = &as[(waveM * 32 + tm * 16 + laneM) * LDSS + laneK * 8];
      v8bf lo = *(const v8bf*)p;
      v8bf hi = *(const v8bf*)(p + 16);
      aF[tm] = __builtin_shufflevector(lo, hi, 0,1,2,3,4,5,6,7,8,9,10,11,12,13,14,15);
    }
    // B fragment 32x16: lane holds column N=lane%16, lane<16 holds K {0..7,16..23}, lane>=16 holds {8..15,24..31}
#pragma unroll
    for (int tn = 0; tn < 4; ++tn) {
      const bf16_t* p = &bs[(waveN * 64 + tn * 16 + laneM) * LDSS + laneK * 8];
      v8bf lo = *(const v8bf*)p;
      v8bf hi = *(const v8bf*)(p + 16);
      bF[tn] = __builtin_shufflevector(lo, hi, 0,1,2,3,4,5,6,7,8,9,10,11,12,13,14,15);
    }
#pragma unroll
    for (int tm = 0; tm < 2; ++tm)
#pragma unroll
      for (int tn = 0; tn < 4; ++tn)
        acc[tm][tn] = __builtin_amdgcn_wmma_f32_16x16x32_bf16(
            false, aF[tm], false, bF[tn], (short)0, acc[tm][tn], false, false);
    cur ^= 1;
  }

  // Epilogue: D layout -> lane row = v + 8*(lane/16), col = lane%16
#pragma unroll
  for (int tm = 0; tm < 2; ++tm) {
#pragma unroll
    for (int tn = 0; tn < 4; ++tn) {
      const size_t col = blockN + waveN * 64 + tn * 16 + laneM;
      const float b = bias ? bias[col] : 0.0f;
#pragma unroll
      for (int v = 0; v < 8; ++v) {
        const size_t row = blockM + waveM * 32 + tm * 16 + laneK * 8 + v;
        const size_t idx = row * (size_t)N + col;
        float val = acc[tm][tn][v] + b;
        if (residual) val += residual[idx];
        C[idx] = val;
      }
    }
  }
}

// ---------------------------------------------------------------------------
// RMSNorm: one block per row; out = gamma * sqrt(dim) * x / ||x||  (bf16 out)
// ---------------------------------------------------------------------------
__global__ __launch_bounds__(256) void rmsnorm_kernel(
    const float* __restrict__ x, const float* __restrict__ gamma,
    bf16_t* __restrict__ out, int dim)
{
  const size_t row = blockIdx.x;
  const float* xr = x + row * (size_t)dim;
  float ss = 0.0f;
  for (int i = threadIdx.x; i < dim; i += 256) { float v = xr[i]; ss += v * v; }
  __shared__ float red[256];
  red[threadIdx.x] = ss;
  __syncthreads();
  for (int s = 128; s > 0; s >>= 1) {
    if (threadIdx.x < s) red[threadIdx.x] += red[threadIdx.x + s];
    __syncthreads();
  }
  const float scale = sqrtf((float)dim / red[0]);
  for (int i = threadIdx.x; i < dim; i += 256)
    out[row * (size_t)dim + i] = (bf16_t)(xr[i] * scale * gamma[i]);
}

// fp32 -> bf16 weight conversion
__global__ __launch_bounds__(256) void f2bf_kernel(
    const float* __restrict__ in, bf16_t* __restrict__ out, int n)
{
  int i = blockIdx.x * 256 + threadIdx.x;
  if (i < n) out[i] = (bf16_t)in[i];
}

// causal depthwise conv K=4 over time on the v-half of u; emits f32 + bf16
__global__ __launch_bounds__(256) void conv_kernel(
    const float* __restrict__ u, const float* __restrict__ cw,
    const float* __restrict__ cb, float* __restrict__ vf, bf16_t* __restrict__ vb)
{
  const size_t idx = (size_t)blockIdx.x * 256 + threadIdx.x;   // over NT*HID
  const int h = (int)(idx % kHid);
  const size_t nt = idx / kHid;
  const int t = (int)(nt % kT);
  const float* src = u + nt * (size_t)kH2 + kHid + h;          // v-half column
  float acc = cb[h];
#pragma unroll
  for (int k = 0; k < 4; ++k) {
    const int dt = t - 3 + k;
    if (dt >= 0) acc += cw[h * 4 + k] * src[(ptrdiff_t)(k - 3) * kH2];
  }
  vf[idx] = acc;
  vb[idx] = (bf16_t)acc;
}

// alpha_t, xg_t from gate projections
__global__ __launch_bounds__(256) void gates_kernel(
    const float* __restrict__ g, const float* __restrict__ fb,
    const float* __restrict__ vf, float* __restrict__ alpha, float* __restrict__ xg)
{
  const size_t idx = (size_t)blockIdx.x * 256 + threadIdx.x;   // over NT*HID
  const int h = (int)(idx % kHid);
  const size_t nt = idx / kHid;
  const float forget = g[nt * (size_t)kH2 + h];
  const float inp    = g[nt * (size_t)kH2 + kHid + h];
  const float sp  = log1pf(expf(fb[h]));                        // softplus(forget_base)
  const float sgf = 1.0f / (1.0f + expf(-forget));
  const float a   = expf(-8.0f * sp * sgf);
  const float beta = sqrtf(1.0f - a * a + 1e-6f);
  const float sgi = 1.0f / (1.0f + expf(-inp));
  alpha[idx] = a;
  xg[idx] = beta * sgi * vf[idx];
}

// scan pass 1: per (n, chunk, h) compute (A = prod alpha, B = local scan end)
__global__ __launch_bounds__(256) void scan_partial(
    const float* __restrict__ alpha, const float* __restrict__ xg,
    float* __restrict__ aprod, float* __restrict__ bend)
{
  const size_t idx = (size_t)blockIdx.x * 256 + threadIdx.x;   // N*NCh*HID
  const int h = (int)(idx % kHid);
  const size_t rest = idx / kHid;
  const int c = (int)(rest % kNCh);
  const int n = (int)(rest / kNCh);
  const size_t base = ((size_t)n * kT + (size_t)c * kCL) * kHid + h;
  float A = 1.0f, B = 0.0f;
  for (int t = 0; t < kCL; ++t) {
    const float a = alpha[base + (size_t)t * kHid];
    const float x = xg[base + (size_t)t * kHid];
    A *= a; B = a * B + x;
  }
  aprod[idx] = A; bend[idx] = B;
}

// scan pass 2: sequential carry combine over the 16 chunks per (n,h)
__global__ __launch_bounds__(256) void scan_carry(
    const float* __restrict__ aprod, const float* __restrict__ bend,
    float* __restrict__ carry)
{
  const size_t idx = (size_t)blockIdx.x * 256 + threadIdx.x;   // N*HID
  const int h = (int)(idx % kHid);
  const int n = (int)(idx / kHid);
  float c = 0.0f;
  for (int ch = 0; ch < kNCh; ++ch) {
    const size_t i = ((size_t)n * kNCh + ch) * kHid + h;
    carry[i] = c;
    c = aprod[i] * c + bend[i];
  }
}

// scan pass 3: replay local scan with carry-in, writing h state
__global__ __launch_bounds__(256) void scan_final(
    const float* __restrict__ alpha, const float* __restrict__ xg,
    const float* __restrict__ carry, float* __restrict__ hout)
{
  const size_t idx = (size_t)blockIdx.x * 256 + threadIdx.x;   // N*NCh*HID
  const int h = (int)(idx % kHid);
  const size_t rest = idx / kHid;
  const int c = (int)(rest % kNCh);
  const int n = (int)(rest / kNCh);
  const size_t base = ((size_t)n * kT + (size_t)c * kCL) * kHid + h;
  float hprev = carry[((size_t)n * kNCh + c) * kHid + h];
  for (int t = 0; t < kCL; ++t) {
    const float a = alpha[base + (size_t)t * kHid];
    const float x = xg[base + (size_t)t * kHid];
    hprev = a * hprev + x;
    hout[base + (size_t)t * kHid] = hprev;
  }
}

// o = gelu_exact(gate) * h  (bf16)
__global__ __launch_bounds__(256) void hawkgate_kernel(
    const float* __restrict__ u, const float* __restrict__ hstate, bf16_t* __restrict__ o)
{
  const size_t idx = (size_t)blockIdx.x * 256 + threadIdx.x;   // NT*HID
  const int h = (int)(idx % kHid);
  const size_t nt = idx / kHid;
  const float gate = u[nt * (size_t)kH2 + h];
  const float ge = 0.5f * gate * (1.0f + erff(gate * 0.70710678118654752f));
  o[idx] = (bf16_t)(ge * hstate[idx]);
}

// m = gelu_exact(gate2) * v2  (bf16)
__global__ __launch_bounds__(256) void mlpgate_kernel(
    const float* __restrict__ grow, bf16_t* __restrict__ m)
{
  const size_t idx = (size_t)blockIdx.x * 256 + threadIdx.x;   // NT*GHID
  const int j = (int)(idx % kGhid);
  const size_t nt = idx / kGhid;
  const float gate = grow[nt * (size_t)kG2 + j];
  const float v2   = grow[nt * (size_t)kG2 + kGhid + j];
  const float ge = 0.5f * gate * (1.0f + erff(gate * 0.70710678118654752f));
  m[idx] = (bf16_t)(ge * v2);
}

// ---------------------------------------------------------------------------
extern "C" void kernel_launch(void* const* d_in, const int* in_sizes, int n_in,
                              void* d_out, int out_size, void* d_ws, size_t ws_size,
                              hipStream_t stream) {
  const float* x        = (const float*)d_in[0];
  const float* gamma1   = (const float*)d_in[1];
  const float* W_in     = (const float*)d_in[2];
  const float* conv_w   = (const float*)d_in[3];
  const float* conv_b   = (const float*)d_in[4];
  const float* W_gates  = (const float*)d_in[5];
  const float* b_gates  = (const float*)d_in[6];
  const float* fbase    = (const float*)d_in[7];
  const float* W_out    = (const float*)d_in[8];
  const float* gamma2   = (const float*)d_in[9];
  const float* W_grow   = (const float*)d_in[10];
  const float* W_shrink = (const float*)d_in[11];
  float* out = (float*)d_out;

  // bump allocator over d_ws
  char* ws = (char*)d_ws;
  size_t off = 0;
  auto alloc = [&](size_t bytes) -> void* {
    void* p = ws + off;
    off = (off + bytes + 255) & ~(size_t)255;
    return p;
  };
  bf16_t* wIn   = (bf16_t*)alloc((size_t)kH2 * kDim * 2);
  bf16_t* wG    = (bf16_t*)alloc((size_t)kH2 * kHid * 2);
  bf16_t* wO    = (bf16_t*)alloc((size_t)kDim * kHid * 2);
  bf16_t* wGr   = (bf16_t*)alloc((size_t)kG2 * kDim * 2);
  bf16_t* wSh   = (bf16_t*)alloc((size_t)kDim * kGhid * 2);
  bf16_t* xn    = (bf16_t*)alloc((size_t)kNT * kDim * 2);
  float*  u     = (float*) alloc((size_t)kNT * kH2 * 4);
  float*  vf    = (float*) alloc((size_t)kNT * kHid * 4);
  bf16_t* vb    = (bf16_t*)alloc((size_t)kNT * kHid * 2);
  float*  g     = (float*) alloc((size_t)kNT * kH2 * 4);
  float*  alpha = (float*) alloc((size_t)kNT * kHid * 4);
  float*  xg    = (float*) alloc((size_t)kNT * kHid * 4);
  float*  aprod = (float*) alloc((size_t)kBatch * kNCh * kHid * 4);
  float*  bend  = (float*) alloc((size_t)kBatch * kNCh * kHid * 4);
  float*  carry = (float*) alloc((size_t)kBatch * kNCh * kHid * 4);
  float*  hst   = (float*) alloc((size_t)kNT * kHid * 4);
  bf16_t* o     = (bf16_t*)alloc((size_t)kNT * kHid * 2);
  float*  x2    = (float*) alloc((size_t)kNT * kDim * 4);
  bf16_t* xn2   = (bf16_t*)alloc((size_t)kNT * kDim * 2);
  float*  grow  = (float*) alloc((size_t)kNT * kG2 * 4);
  bf16_t* m     = (bf16_t*)alloc((size_t)kNT * kGhid * 2);

  auto cvt = [&](const float* src, bf16_t* dst, int n) {
    f2bf_kernel<<<(n + 255) / 256, 256, 0, stream>>>(src, dst, n);
  };
  cvt(W_in,     wIn, kH2 * kDim);
  cvt(W_gates,  wG,  kH2 * kHid);
  cvt(W_out,    wO,  kDim * kHid);
  cvt(W_grow,   wGr, kG2 * kDim);
  cvt(W_shrink, wSh, kDim * kGhid);

  auto gemm = [&](const bf16_t* A, const bf16_t* W, float* C, int M, int N, int K,
                  const float* bias, const float* res) {
    dim3 grid(N / BN, M / BM);
    gemm_bf16<<<grid, 256, 0, stream>>>(A, W, C, M, N, K, bias, res);
  };

  const int eHID  = kNT * kHid;    // 12.6M
  const int eGHID = kNT * kGhid;   // 16.8M

  // --- Hawk branch ---
  rmsnorm_kernel<<<kNT, 256, 0, stream>>>(x, gamma1, xn, kDim);
  gemm(xn, wIn, u, kNT, kH2, kDim, nullptr, nullptr);                 // u = xn @ W_in^T
  conv_kernel<<<eHID / 256, 256, 0, stream>>>(u, conv_w, conv_b, vf, vb);
  gemm(vb, wG, g, kNT, kH2, kHid, b_gates, nullptr);                  // g = v @ W_gates^T + b
  gates_kernel<<<eHID / 256, 256, 0, stream>>>(g, fbase, vf, alpha, xg);
  scan_partial<<<(kBatch * kNCh * kHid) / 256, 256, 0, stream>>>(alpha, xg, aprod, bend);
  scan_carry<<<(kBatch * kHid) / 256, 256, 0, stream>>>(aprod, bend, carry);
  scan_final<<<(kBatch * kNCh * kHid) / 256, 256, 0, stream>>>(alpha, xg, carry, hst);
  hawkgate_kernel<<<eHID / 256, 256, 0, stream>>>(u, hst, o);
  gemm(o, wO, x2, kNT, kDim, kHid, nullptr, x);                       // x2 = x + hawk_out

  // --- GatedMLP branch ---
  rmsnorm_kernel<<<kNT, 256, 0, stream>>>(x2, gamma2, xn2, kDim);
  gemm(xn2, wGr, grow, kNT, kG2, kDim, nullptr, nullptr);             // grow
  mlpgate_kernel<<<eGHID / 256, 256, 0, stream>>>(grow, m);
  gemm(m, wSh, out, kNT, kDim, kGhid, nullptr, x2);                   // out = x2 + mlp
}
